// MultiScaleAttention_28578712388187
// MI455X (gfx1250) — matrix operand order repaired and checked by reading, not compiled, not run on hardware
//
#include <hip/hip_runtime.h>
#include <hip/hip_bf16.h>

// ---------------------------------------------------------------------------
// MultiScaleAttention (PVT-style SRA) for gfx1250, bf16 WMMA + flash softmax.
// B=2, N=4096, C=768, H=12, d=64, pool 3/3/pad1 -> Np=1366 (pad to 1376).
// GEMM waves: 32x64 register tiles (8 wmma / K-step) + global_prefetch_b8.
// Attention: 8 waves/block share one head; K/V tiles DMA'd by the Tensor Data
// Mover into LDS, double-buffered against the WMMA pipeline (TENSORcnt).
// ---------------------------------------------------------------------------

typedef __bf16 bf16;
typedef bf16  v16bf __attribute__((ext_vector_type(16)));
typedef bf16  v8bf  __attribute__((ext_vector_type(8)));
typedef float v8f   __attribute__((ext_vector_type(8)));
typedef unsigned int u32x4 __attribute__((ext_vector_type(4)));
typedef int i32x8 __attribute__((ext_vector_type(8)));
typedef int i32x4 __attribute__((ext_vector_type(4)));

#define B_   2
#define N_   4096
#define C_   768
#define H_   12
#define D_   64
#define NP_  1366
#define NPP_ 1376           // padded pooled length (multiple of 32)
#define M_   (B_ * N_)      // 8192 rows
#define NITER_ (NPP_ / 32)  // 43 attention steps

// ---------------- fragment loaders (contiguous-k 16-bit layouts) -------------

// A matrix 16x32 (MxK), row-major source with leading dim lda.
// lanes 0-15: M=lane, K = 0..15; lanes 16-31: M=lane-16, K = 16..31
__device__ __forceinline__ v16bf load_a16x32(const bf16* A, int lda, int row0,
                                             int k0, int lane) {
  int r    = row0 + (lane & 15);
  int koff = (lane >> 4) << 4;
  const bf16* p = A + (size_t)r * lda + k0 + koff;
  union { v16bf v; v8bf h[2]; } u;
  u.h[0] = *(const v8bf*)(p);
  u.h[1] = *(const v8bf*)(p + 8);
  return u.v;
}

// B matrix 32x16 (KxN) from a TRANSPOSED source Bt[N, ldb] (row n = column of B).
// lanes 0-15: N=lane, K=0..15 ; lanes 16-31: N=lane-16, K=16..31 (2 K per VGPR)
__device__ __forceinline__ v16bf load_bt32x16(const bf16* Bt, int ldb, int n0,
                                              int k0, int lane) {
  int n    = n0 + (lane & 15);
  int koff = (lane >> 4) << 4;
  const bf16* p = Bt + (size_t)n * ldb + k0 + koff;
  return *(const v16bf*)p;   // 32 contiguous bytes
}

__device__ __forceinline__ v8f wmma_bf16(v16bf a, v16bf b, v8f c) {
  return __builtin_amdgcn_wmma_f32_16x16x32_bf16(false, a, false, b, (short)0, c,
                                                 false, false);
}

// ---------------- Tensor Data Mover: 1-D contiguous bf16 copy to LDS --------
// D# per CDNA5 ISA 8.3/8.4: group0 {count=1, lds_addr, global_addr, type=2},
// group1 {data_size=2B, tensor_dim0=tile_dim0=nelem, tensor_dim1=tile_dim1=1},
// groups 2/3 zero (<=2D tensor). 6-arg clang-23 builtin form.

__device__ __forceinline__ unsigned lds_off(const void* p) {
  return (unsigned)(uintptr_t)p;   // LDS aperture: low 32 bits = LDS byte addr
}

__device__ __forceinline__ void tdm_copy_bf16(const void* gsrc, unsigned lds_addr,
                                              unsigned nelem) {
  unsigned long long ga = (unsigned long long)(uintptr_t)gsrc;
  u32x4 g0;
  g0.x = 0x1u;                                        // count=1 (valid user D#)
  g0.y = lds_addr;                                    // LDS byte address
  g0.z = (unsigned)(ga & 0xFFFFFFFFu);                // global_addr[31:0]
  g0.w = (unsigned)((ga >> 32) & 0x01FFFFFFu) | 0x80000000u; // addr[56:32]|type=2
  i32x8 g1;
  g1[0] = 0x00010000;                                 // data_size = 1 -> 2 bytes
  g1[1] = (int)((nelem & 0xFFFFu) << 16);             // tensor_dim0[15:0]
  g1[2] = (int)(((nelem >> 16) & 0xFFFFu) | (1u << 16)); // dim0 hi | tensor_dim1=1
  g1[3] = (int)((nelem & 0xFFFFu) << 16);             // tile_dim0 = nelem
  g1[4] = 1;                                          // tile_dim1 = 1
  g1[5] = (int)nelem;                                 // tensor_dim0_stride
  g1[6] = 0;
  g1[7] = 0;
  i32x4 z4 = {0, 0, 0, 0};
  i32x8 z8 = {0, 0, 0, 0, 0, 0, 0, 0};
  __builtin_amdgcn_tensor_load_to_lds(g0, g1, z4, z4, z8, 0);
}

// ---------------- conversion kernels ---------------------------------------

__global__ void msa_cvt_bf16(const float* __restrict__ src, bf16* __restrict__ dst,
                             int n) {
  int i = blockIdx.x * 256 + threadIdx.x;
  if (i < n) dst[i] = (bf16)src[i];
}

// src [rows, cols] f32  ->  dst [cols, rows] bf16
__global__ void msa_cvt_transpose(const float* __restrict__ src, bf16* __restrict__ dst,
                                  int rows, int cols) {
  int i = blockIdx.x * 256 + threadIdx.x;
  if (i < rows * cols) {
    int r = i / cols, c = i - r * cols;
    dst[(size_t)c * rows + r] = (bf16)src[i];
  }
}

// ---------------- QKV GEMM: [8192,768] @ [768,2304] -------------------------

__global__ __launch_bounds__(256)
void msa_gemm_qkv(const bf16* __restrict__ A, const bf16* __restrict__ Bt,
                  bf16* __restrict__ qb, bf16* __restrict__ kb,
                  bf16* __restrict__ vb) {
  int lane = threadIdx.x & 31;
  int wv   = threadIdx.x >> 5;
  int wid  = blockIdx.x * 8 + wv;       // 0..9215
  int tn   = wid % 36;                  // 64-wide col tiles over 2304
  int tm   = wid / 36;                  // 32-high row tiles over 8192
  int n0   = tn * 64;
  int m0   = tm * 32;
  int l15  = lane & 15, hi = lane >> 4;

  v8f acc[2][4];
#pragma unroll
  for (int mi = 0; mi < 2; ++mi)
#pragma unroll
    for (int j = 0; j < 4; ++j) acc[mi][j] = (v8f){};

#pragma unroll 2
  for (int k0 = 0; k0 < C_; k0 += 32) {
    if (k0 + 32 < C_) {
      __builtin_prefetch(A + (size_t)(m0 + l15) * C_ + k0 + 32, 0, 0);
      __builtin_prefetch(Bt + (size_t)(n0 + l15) * C_ + k0 + 32, 0, 0);
    }
    v16bf a0 = load_a16x32(A, C_, m0, k0, lane);
    v16bf a1 = load_a16x32(A, C_, m0 + 16, k0, lane);
#pragma unroll
    for (int j = 0; j < 4; ++j) {
      v16bf b = load_bt32x16(Bt, C_, n0 + 16 * j, k0, lane);
      acc[0][j] = wmma_bf16(a0, b, acc[0][j]);
      acc[1][j] = wmma_bf16(a1, b, acc[1][j]);
    }
  }

  // n0 is a multiple of 64 -> a single (qkv, head) per tile
  int qkv = n0 / C_;
  int rem = n0 - qkv * C_;
  int h   = rem >> 6;
  bf16* dst = (qkv == 0) ? qb : ((qkv == 1) ? kb : vb);
  float mul = (qkv == 0) ? 0.125f : 1.0f;   // 64^-0.5

#pragma unroll
  for (int mi = 0; mi < 2; ++mi) {
#pragma unroll
    for (int r = 0; r < 8; ++r) {
      int m = m0 + 16 * mi + r + (hi << 3);
      int bb = m >> 12, n = m & (N_ - 1);
      size_t base = (((size_t)(bb * H_ + h) * N_ + n) << 6) + l15;
#pragma unroll
      for (int j = 0; j < 4; ++j)
        dst[base + 16 * j] = (bf16)(acc[mi][j][r] * mul);
    }
  }
}

// ---------------- pooling: k,v [B,H,4096,64] -> kp [B,H,1376,64],
// vp in 16-row-chunk transposed layout: vp[bh][m>>4][d][m&15] ----------------

__global__ void msa_pool(const bf16* __restrict__ kb, const bf16* __restrict__ vb,
                         bf16* __restrict__ kp, bf16* __restrict__ vp) {
  int t = blockIdx.x * 256 + threadIdx.x;       // B*H*NPP*64 threads
  if (t >= B_ * H_ * NPP_ * D_) return;
  int d  = t & 63;
  int m  = (t >> 6) % NPP_;
  int bh = t / (NPP_ * D_);

  float sk = 0.f, sv = 0.f;
  if (m < NP_) {
    int nb = 3 * m - 1;
#pragma unroll
    for (int j = 0; j < 3; ++j) {
      int n = nb + j;
      if (n >= 0 && n < N_) {
        size_t idx = (((size_t)bh * N_ + n) << 6) + d;
        sk += (float)kb[idx];
        sv += (float)vb[idx];
      }
    }
    sk *= (1.f / 3.f);
    sv *= (1.f / 3.f);
  }
  kp[(((size_t)bh * NPP_ + m) << 6) + d] = (bf16)sk;
  vp[((size_t)bh * (NPP_ / 16) + (m >> 4)) * 1024 + d * 16 + (m & 15)] = (bf16)sv;
}

// ---------------- flash attention -------------------------------------------
// Block = 8 waves sharing one (b,h); each wave owns 32 q-rows (256 rows/block).
// Per 32-column step: K tile (32x64) and V chunk-pair (2x1024) are TDM-DMA'd
// into double-buffered LDS by wave 0, overlapped with 16 WMMAs/wave; barriers
// publish buffers. Online softmax via half-wave shfl_xor; P bounced through
// per-wave LDS to become A-fragments.

__global__ __launch_bounds__(256)
void msa_attn(const bf16* __restrict__ qb, const bf16* __restrict__ kp,
              const bf16* __restrict__ vp, bf16* __restrict__ ab) {
  __shared__ bf16 kbuf[2][32 * 64];     // 2 x 4KB
  __shared__ bf16 vbuf[2][2 * 1024];    // 2 x 4KB
  __shared__ bf16 plds[8][32 * 32];     // 8 x 2KB

  int lane = threadIdx.x & 31;
  int wv   = threadIdx.x >> 5;
  int bh   = blockIdx.x >> 4;             // 0..23
  int qblk = blockIdx.x & 15;             // 16 blocks of 256 rows over N
  int n0   = qblk * 256 + wv * 32;
  int l15  = lane & 15, hi = lane >> 4;

  const bf16* qbase = qb + ((size_t)bh << 18);          // *4096*64
  const bf16* kbase = kp + (size_t)bh * (NPP_ * D_);
  const bf16* vbase = vp + (size_t)bh * ((NPP_ / 16) * 1024);

  // prologue: DMA iteration 0 tiles into buffer 0
  if (wv == 0) {
    tdm_copy_bf16(kbase, lds_off(&kbuf[0][0]), 32 * 64);
    tdm_copy_bf16(vbase, lds_off(&vbuf[0][0]), 2 * 1024);
  }

  v16bf qa[2][2];
#pragma unroll
  for (int mi = 0; mi < 2; ++mi) {
    qa[mi][0] = load_a16x32(qbase, D_, n0 + 16 * mi, 0, lane);
    qa[mi][1] = load_a16x32(qbase, D_, n0 + 16 * mi, 32, lane);
  }

  v8f o[2][4];
  float lsum[2][8], mrow[2][8];
#pragma unroll
  for (int mi = 0; mi < 2; ++mi) {
#pragma unroll
    for (int j = 0; j < 4; ++j) o[mi][j] = (v8f){};
#pragma unroll
    for (int r = 0; r < 8; ++r) { lsum[mi][r] = 0.f; mrow[mi][r] = -__builtin_inff(); }
  }

  for (int it = 0; it < NITER_; ++it) {
    int m0 = it * 32;
    __syncthreads();   // all waves done with buffer (it+1)&1 from step it-1
    if (wv == 0) {
      if (it + 1 < NITER_) {
        int nb = (it + 1) & 1;
        tdm_copy_bf16(kbase + (size_t)(m0 + 32) * D_, lds_off(&kbuf[nb][0]), 32 * 64);
        tdm_copy_bf16(vbase + (size_t)((m0 + 32) >> 4) * 1024, lds_off(&vbuf[nb][0]),
                      2 * 1024);
        __builtin_amdgcn_s_wait_tensorcnt(2);   // step-it transfers complete
      } else {
        __builtin_amdgcn_s_wait_tensorcnt(0);
      }
    }
    __syncthreads();   // publish buffer it&1
    const bf16* kt = &kbuf[it & 1][0];
    const bf16* vt = &vbuf[it & 1][0];

    // ---- shared K fragments (from LDS)
    v16bf kb00 = load_bt32x16(kt, D_, 0, 0, lane);
    v16bf kb01 = load_bt32x16(kt, D_, 0, 32, lane);
    v16bf kb10 = load_bt32x16(kt, D_, 16, 0, lane);
    v16bf kb11 = load_bt32x16(kt, D_, 16, 32, lane);
    // ---- shared V fragments (chunk-transposed layout -> contiguous 32B)
    const bf16* vchunk = vt + hi * 1024;
    v16bf vb0 = *(const v16bf*)(vchunk + (0  + l15) * 16);
    v16bf vb1 = *(const v16bf*)(vchunk + (16 + l15) * 16);
    v16bf vb2 = *(const v16bf*)(vchunk + (32 + l15) * 16);
    v16bf vb3 = *(const v16bf*)(vchunk + (48 + l15) * 16);

    bool mask0 = (m0 + l15) >= NP_;
    bool mask1 = (m0 + 16 + l15) >= NP_;

#pragma unroll
    for (int mi = 0; mi < 2; ++mi) {
      v8f s0 = {}, s1 = {};
      s0 = wmma_bf16(qa[mi][0], kb00, s0);
      s0 = wmma_bf16(qa[mi][1], kb01, s0);
      s1 = wmma_bf16(qa[mi][0], kb10, s1);
      s1 = wmma_bf16(qa[mi][1], kb11, s1);
      if (mask0) {
#pragma unroll
        for (int r = 0; r < 8; ++r) s0[r] = -1.0e30f;
      }
      if (mask1) {
#pragma unroll
        for (int r = 0; r < 8; ++r) s1[r] = -1.0e30f;
      }
      // ---- online softmax (rows live across 16 lanes of a half-wave)
      float alpha[8];
#pragma unroll
      for (int r = 0; r < 8; ++r) {
        float mx = fmaxf(s0[r], s1[r]);
        mx = fmaxf(mx, __shfl_xor(mx, 1, 32));
        mx = fmaxf(mx, __shfl_xor(mx, 2, 32));
        mx = fmaxf(mx, __shfl_xor(mx, 4, 32));
        mx = fmaxf(mx, __shfl_xor(mx, 8, 32));
        float mn = fmaxf(mrow[mi][r], mx);
        alpha[r]    = __expf(mrow[mi][r] - mn);
        mrow[mi][r] = mn;
        float p0 = __expf(s0[r] - mn);
        float p1 = __expf(s1[r] - mn);
        int row = 16 * mi + r + (hi << 3);
        plds[wv][row * 32 + l15]      = (bf16)p0;
        plds[wv][row * 32 + 16 + l15] = (bf16)p1;
        float rs = p0 + p1;
        rs += __shfl_xor(rs, 1, 32);
        rs += __shfl_xor(rs, 2, 32);
        rs += __shfl_xor(rs, 4, 32);
        rs += __shfl_xor(rs, 8, 32);
        lsum[mi][r] = lsum[mi][r] * alpha[r] + rs;
      }
      // ---- rescale running output for this row tile
#pragma unroll
      for (int r = 0; r < 8; ++r) {
        o[mi][0][r] *= alpha[r];
        o[mi][1][r] *= alpha[r];
        o[mi][2][r] *= alpha[r];
        o[mi][3][r] *= alpha[r];
      }
    }

    // ---- P A-fragments from LDS (C layout -> A layout transpose)
#pragma unroll
    for (int mi = 0; mi < 2; ++mi) {
      union { v16bf v; v8bf h[2]; } pu;
      const bf16* pp = &plds[wv][(16 * mi + l15) * 32 + (hi << 4)];
      pu.h[0] = *(const v8bf*)(pp);
      pu.h[1] = *(const v8bf*)(pp + 8);
      o[mi][0] = wmma_bf16(pu.v, vb0, o[mi][0]);
      o[mi][1] = wmma_bf16(pu.v, vb1, o[mi][1]);
      o[mi][2] = wmma_bf16(pu.v, vb2, o[mi][2]);
      o[mi][3] = wmma_bf16(pu.v, vb3, o[mi][3]);
    }
  }

  // ---- epilogue: out[b, n, h*64 + d] bf16 for the projection GEMM
  int bb = bh / H_, hh = bh - bb * H_;
#pragma unroll
  for (int mi = 0; mi < 2; ++mi) {
#pragma unroll
    for (int r = 0; r < 8; ++r) {
      int n = n0 + 16 * mi + r + (hi << 3);
      float rl = 1.0f / lsum[mi][r];
      size_t base = ((size_t)(bb * N_ + n)) * C_ + hh * 64 + l15;
      ab[base + 0]  = (bf16)(o[mi][0][r] * rl);
      ab[base + 16] = (bf16)(o[mi][1][r] * rl);
      ab[base + 32] = (bf16)(o[mi][2][r] * rl);
      ab[base + 48] = (bf16)(o[mi][3][r] * rl);
    }
  }
}

// ---------------- projection GEMM: [8192,768] @ [768,768] + bias -> f32 -----

__global__ __launch_bounds__(256)
void msa_gemm_proj(const bf16* __restrict__ A, const bf16* __restrict__ Bt,
                   const float* __restrict__ bias, float* __restrict__ out) {
  int lane = threadIdx.x & 31;
  int wv   = threadIdx.x >> 5;
  int wid  = blockIdx.x * 8 + wv;   // 0..3071
  int tn   = wid % 12;              // 64-wide col tiles over 768
  int tm   = wid / 12;              // 32-high row tiles over 8192
  int n0   = tn * 64;
  int m0   = tm * 32;
  int l15  = lane & 15, hi = lane >> 4;

  v8f acc[2][4];
#pragma unroll
  for (int mi = 0; mi < 2; ++mi)
#pragma unroll
    for (int j = 0; j < 4; ++j) acc[mi][j] = (v8f){};

#pragma unroll 2
  for (int k0 = 0; k0 < C_; k0 += 32) {
    if (k0 + 32 < C_) {
      __builtin_prefetch(A + (size_t)(m0 + l15) * C_ + k0 + 32, 0, 0);
      __builtin_prefetch(Bt + (size_t)(n0 + l15) * C_ + k0 + 32, 0, 0);
    }
    v16bf a0 = load_a16x32(A, C_, m0, k0, lane);
    v16bf a1 = load_a16x32(A, C_, m0 + 16, k0, lane);
#pragma unroll
    for (int j = 0; j < 4; ++j) {
      v16bf b = load_bt32x16(Bt, C_, n0 + 16 * j, k0, lane);
      acc[0][j] = wmma_bf16(a0, b, acc[0][j]);
      acc[1][j] = wmma_bf16(a1, b, acc[1][j]);
    }
  }

  float bs[4];
#pragma unroll
  for (int j = 0; j < 4; ++j) bs[j] = bias[n0 + 16 * j + l15];

#pragma unroll
  for (int mi = 0; mi < 2; ++mi) {
#pragma unroll
    for (int r = 0; r < 8; ++r) {
      int m = m0 + 16 * mi + r + (hi << 3);
      size_t base = (size_t)m * C_ + n0 + l15;
#pragma unroll
      for (int j = 0; j < 4; ++j)
        out[base + 16 * j] = acc[mi][j][r] + bs[j];
    }
  }
}

// ---------------- host-side launch ------------------------------------------

extern "C" void kernel_launch(void* const* d_in, const int* in_sizes, int n_in,
                              void* d_out, int out_size, void* d_ws, size_t ws_size,
                              hipStream_t stream) {
  const float* x      = (const float*)d_in[0];   // [2,4096,768]
  const float* w_qkv  = (const float*)d_in[1];   // [768,2304]
  const float* w_proj = (const float*)d_in[2];   // [768,768]
  const float* b_proj = (const float*)d_in[3];   // [768]
  float*       out    = (float*)d_out;           // [2,4096,768]

  char* ws = (char*)d_ws;
  // byte offsets (all 256-aligned); total ~72.6 MB
  bf16* x_bf   = (bf16*)(ws + 0);                 // 8192*768*2   = 12,582,912
  bf16* wqkvT  = (bf16*)(ws + 12582912);          // 2304*768*2   =  3,538,944
  bf16* wprojT = (bf16*)(ws + 16121856);          // 768*768*2    =  1,179,648
  bf16* q_bf   = (bf16*)(ws + 17301504);          // 12,582,912
  bf16* k_bf   = (bf16*)(ws + 29884416);          // 12,582,912
  bf16* v_bf   = (bf16*)(ws + 42467328);          // 12,582,912
  bf16* kp_bf  = (bf16*)(ws + 55050240);          // 2*12*1376*64*2 = 4,227,072
  bf16* vp_bf  = (bf16*)(ws + 59277312);          // 4,227,072
  bf16* a_bf   = (bf16*)(ws + 63504384);          // 12,582,912

  const int nx  = M_ * C_;                // 6,291,456
  const int nwq = C_ * 3 * C_;            // 1,769,472
  const int nwp = C_ * C_;                // 589,824

  msa_cvt_bf16<<<(nx + 255) / 256, 256, 0, stream>>>(x, x_bf, nx);
  msa_cvt_transpose<<<(nwq + 255) / 256, 256, 0, stream>>>(w_qkv, wqkvT, C_, 3 * C_);
  msa_cvt_transpose<<<(nwp + 255) / 256, 256, 0, stream>>>(w_proj, wprojT, C_, C_);

  // 8192/32 * 2304/64 = 256*36 = 9216 waves -> 1152 blocks
  msa_gemm_qkv<<<1152, 256, 0, stream>>>(x_bf, wqkvT, q_bf, k_bf, v_bf);

  const int npool = B_ * H_ * NPP_ * D_;  // 2,113,536
  msa_pool<<<(npool + 255) / 256, 256, 0, stream>>>(k_bf, v_bf, kp_bf, vp_bf);

  // 24 heads*batches x 16 row-blocks = 384 blocks, 8 waves each
  msa_attn<<<384, 256, 0, stream>>>(q_bf, kp_bf, vp_bf, a_bf);

  // 8192/32 * 768/64 = 256*12 = 3072 waves -> 384 blocks
  msa_gemm_proj<<<384, 256, 0, stream>>>(a_bf, wprojT, b_proj, out);
}
